// AREConvFPN_14096082665966
// MI455X (gfx1250) — matrix operand order repaired and checked by reading, not compiled, not run on hardware
//
#include <hip/hip_runtime.h>
#include <hip/hip_bf16.h>

#define EPSV      1e-6f
#define NEGS      0.2f
#define BNS       0.99999500003749937f   /* 1/sqrt(1+1e-5) */

typedef __attribute__((ext_vector_type(2)))  float    v2f;
typedef __attribute__((ext_vector_type(8)))  float    v8f;
typedef __attribute__((ext_vector_type(16))) _Float16 v16h;

#ifndef __has_builtin
#define __has_builtin(x) 0
#endif
#if __has_builtin(__builtin_amdgcn_wmma_f32_16x16x4_f32)
#define USE_WMMA_F32 1
#else
#define USE_WMMA_F32 0
#endif

// ---------------------------------------------------------------------------
// Wave-level 16x16 f32 tile GEMM, K=32 (8 steps of V_WMMA_F32_16X16X4_F32).
// S: LDS activations [rows x ss] (ss odd -> conflict-free A fetch),
// W: LDS weights [32 x ws_] (2*ws_ == 16 mod 64 -> halves on disjoint banks).
// ---------------------------------------------------------------------------
__device__ __forceinline__ v8f wmma_tile_k32(const float* S, int ss, int row0,
                                             const float* W, int ws_, int col0)
{
    const int lane = threadIdx.x & 31;
    const int hf   = lane >> 4;     // 0: lanes 0-15, 1: lanes 16-31
    const int l    = lane & 15;
    v8f acc = {0.f, 0.f, 0.f, 0.f, 0.f, 0.f, 0.f, 0.f};
#if USE_WMMA_F32
    #pragma unroll
    for (int kk = 0; kk < 32; kk += 4) {
        // A 16x4 f32: lanes0-15 carry K=kk..kk+1, lanes16-31 K=kk+2..kk+3
        v2f a, b;
        a[0] = S[(row0 + l) * ss + kk + 2 * hf + 0];
        a[1] = S[(row0 + l) * ss + kk + 2 * hf + 1];
        // B 4x16 f32: VGPR v, lanes0-15 -> row kk+v ; lanes16-31 -> row kk+2+v
        b[0] = W[(kk + 2 * hf + 0) * ws_ + col0 + l];
        b[1] = W[(kk + 2 * hf + 1) * ws_ + col0 + l];
        acc = __builtin_amdgcn_wmma_f32_16x16x4_f32(false, a, false, b,
                                                    (short)0, acc, false, false);
    }
#else
    v16h a, b;
    #pragma unroll
    for (int i = 0; i < 8; ++i) {
        a[i]     = (_Float16)S[(row0 + l) * ss + 8 * hf + i];
        a[8 + i] = (_Float16)S[(row0 + l) * ss + 16 + 8 * hf + i];
    }
    #pragma unroll
    for (int i = 0; i < 16; ++i)
        b[i] = (_Float16)W[((i < 8 ? 0 : 8) + 8 * hf + i) * ws_ + col0 + l];
    acc = __builtin_amdgcn_wmma_f32_16x16x32_f16(false, a, false, b,
                                                 (short)0, acc, false, false);
#endif
    return acc;
}

__device__ __forceinline__ void vn_act3(const float* pv, const float* dv, float* ov)
{
    float dot = pv[0] * dv[0] + pv[1] * dv[1] + pv[2] * dv[2];
    float dsq = dv[0] * dv[0] + dv[1] * dv[1] + dv[2] * dv[2];
    float f   = dot / (dsq + EPSV);
    #pragma unroll
    for (int d = 0; d < 3; ++d) {
        float corr = pv[d] - f * dv[d];
        float sel  = (dot >= 0.f) ? pv[d] : corr;
        ov[d] = NEGS * pv[d] + (1.f - NEGS) * sel;
    }
}

// ---------------------------------------------------------------------------
// Kernel A: gather + local frame + score net + weighted KP-conv + normalize
// + mean over K.  16 lanes per point (lane == neighbor k), 16 points / block.
// Writes featsT[n][d(0..2)][c(0..31)] (c contiguous for the tail GEMM rows).
// ---------------------------------------------------------------------------
__global__ __launch_bounds__(256)
void feats_kernel(const float* __restrict__ q, const float* __restrict__ sp,
                  const int* __restrict__ nbr,
                  const float* __restrict__ w_vn, const float* __restrict__ wd_vn,
                  const float* __restrict__ w_h1, const float* __restrict__ w_h2,
                  const float* __restrict__ b_h2, const float* __restrict__ wb,
                  float* __restrict__ featsT, int N)
{
    __shared__ float Lwvn[48], Lwdvn[48], Lwh1[128], Lwh2[64], Lb[8], Lwb[768];
    const int tid = threadIdx.x;
    for (int i = tid; i < 48;  i += 256) { Lwvn[i] = w_vn[i]; Lwdvn[i] = wd_vn[i]; }
    for (int i = tid; i < 128; i += 256) Lwh1[i] = w_h1[i];
    for (int i = tid; i < 64;  i += 256) Lwh2[i] = w_h2[i];
    for (int i = tid; i < 8;   i += 256) Lb[i]   = b_h2[i];
    for (int i = tid; i < 768; i += 256) Lwb[i]  = wb[i];
    __syncthreads();

    const int n = blockIdx.x * 16 + (tid >> 4);
    const int k = tid & 15;
    if (n >= N) return;

    // gather + center + cross -> local frame channels (pt, ct, cr)
    const int idx = nbr[n * 16 + k];
    float pt[3], ct[3], cr[3];
    #pragma unroll
    for (int d = 0; d < 3; ++d) pt[d] = sp[idx * 3 + d] - q[n * 3 + d];
    #pragma unroll
    for (int d = 0; d < 3; ++d) {
        float s = pt[d];
        #pragma unroll
        for (int m = 1; m < 16; m <<= 1) s += __shfl_xor(s, m, 16);
        ct[d] = s * (1.f / 16.f);
    }
    cr[0] = pt[1] * ct[2] - pt[2] * ct[1];
    cr[1] = pt[2] * ct[0] - pt[0] * ct[2];
    cr[2] = pt[0] * ct[1] - pt[1] * ct[0];

    // VNLinearLeakyReLU(3 -> 16) + vector norm  -> sarr[16]
    float sarr[16];
    #pragma unroll
    for (int o = 0; o < 16; ++o) {
        float pv[3], dv[3], dot = 0.f, dsq = 0.f;
        #pragma unroll
        for (int d = 0; d < 3; ++d) {
            float p_ = (pt[d] * Lwvn[o] + ct[d] * Lwvn[16 + o] + cr[d] * Lwvn[32 + o]) * BNS;
            float d_ =  pt[d] * Lwdvn[o] + ct[d] * Lwdvn[16 + o] + cr[d] * Lwdvn[32 + o];
            pv[d] = p_; dv[d] = d_;
            dot += p_ * d_; dsq += d_ * d_;
        }
        float f = dot / (dsq + EPSV), ssum = 0.f;
        #pragma unroll
        for (int d = 0; d < 3; ++d) {
            float corr = pv[d] - f * dv[d];
            float sel  = (dot >= 0.f) ? pv[d] : corr;
            float v    = NEGS * pv[d] + (1.f - NEGS) * sel;
            ssum += v * v;
        }
        sarr[o] = __builtin_sqrtf(ssum);
    }

    // 16 -> 8 (BN+relu), 8 -> 8 (+bias), softmax over the 8 score channels
    float h[8], z[8], sc[8];
    #pragma unroll
    for (int j = 0; j < 8; ++j) {
        float t = 0.f;
        #pragma unroll
        for (int o = 0; o < 16; ++o) t += sarr[o] * Lwh1[o * 8 + j];
        h[j] = fmaxf(t * BNS, 0.f);
    }
    #pragma unroll
    for (int j = 0; j < 8; ++j) {
        float t = Lb[j];
        #pragma unroll
        for (int i = 0; i < 8; ++i) t += h[i] * Lwh2[i * 8 + j];
        z[j] = t;
    }
    float mx = z[0];
    #pragma unroll
    for (int j = 1; j < 8; ++j) mx = fmaxf(mx, z[j]);
    float es = 0.f;
    #pragma unroll
    for (int j = 0; j < 8; ++j) { sc[j] = __expf(z[j] - mx); es += sc[j]; }
    float inv = 1.f / es;
    #pragma unroll
    for (int j = 0; j < 8; ++j) sc[j] *= inv;

    // score-weighted kernel-point conv (fused: w_eff = sum_s sc[s]*wb[:,s,:]),
    // per-k L2 normalize over d, mean over k
    for (int hh = 0; hh < 32; ++hh) {
        float wc0 = 0.f, wc1 = 0.f, wc2 = 0.f;
        #pragma unroll
        for (int s8 = 0; s8 < 8; ++s8) {
            float sv = sc[s8];
            wc0 += sv * Lwb[      s8 * 32 + hh];
            wc1 += sv * Lwb[256 + s8 * 32 + hh];
            wc2 += sv * Lwb[512 + s8 * 32 + hh];
        }
        float pr[3], nn = 0.f;
        #pragma unroll
        for (int d = 0; d < 3; ++d) {
            pr[d] = pt[d] * wc0 + ct[d] * wc1 + cr[d] * wc2;
            nn += pr[d] * pr[d];
        }
        float invn = 1.f / fmaxf(__builtin_sqrtf(nn), 1e-12f);
        #pragma unroll
        for (int d = 0; d < 3; ++d) {
            float v = pr[d] * invn;
            #pragma unroll
            for (int m = 1; m < 16; m <<= 1) v += __shfl_xor(v, m, 16);
            if (k == 0) featsT[n * 96 + d * 32 + hh] = v * (1.f / 16.f);
        }
    }
}

// ---------------------------------------------------------------------------
// Kernel B: WMMA tail.  4 waves/block, 16 points/wave (64 points/block).
// Rows are point-padded: row = 4p + d (d==3 is a zero pad row), so in the
// WMMA D layout every lane holds complete d-triples in acc[0..2]/acc[4..6]
// -> both vn_act epilogues are register-resident (no C-matrix LDS staging).
//   G1 = S1 @ wd_relu  (reg)         [64x32] = [64x32][32x32]
//   S1 <- vn_act(S1, G1)             in place
//   out = vn_act(BNS*(S1@w_un), S1@wd_un)  stored straight to global
// ---------------------------------------------------------------------------
#define TS1 33   /* activation row stride (odd -> conflict-free A fetch)   */
#define TW1 40   /* 32-col weight stride: 2*40 % 64 == 16 -> disjoint banks */
#define TW2 72   /* 64-col weight stride: 2*72 % 64 == 16 -> disjoint banks */

__global__ __launch_bounds__(128)
void tail_kernel(const float* __restrict__ featsT,
                 const float* __restrict__ wd_relu,
                 const float* __restrict__ w_un,
                 const float* __restrict__ wd_un,
                 float* __restrict__ out, int N)
{
    __shared__ float Wr [32 * TW1];
    __shared__ float Wu [32 * TW2];
    __shared__ float Wdu[32 * TW2];
    __shared__ float S1 [4][64 * TS1];   // per-wave activation buffer

    const int tid  = threadIdx.x;
    const int wid  = tid >> 5;
    const int lane = tid & 31;
    const int hf   = lane >> 4;
    const int l    = lane & 15;
    const int n0w  = blockIdx.x * 64 + wid * 16;   // first point of this wave

    // block-cooperative weight staging
    for (int i = tid; i < 32 * 32; i += 128)
        Wr[(i >> 5) * TW1 + (i & 31)] = wd_relu[i];
    for (int i = tid; i < 32 * 64; i += 128) {
        int r = i >> 6, c = i & 63;
        Wu [r * TW2 + c] = w_un [i];
        Wdu[r * TW2 + c] = wd_un[i];
    }
    // per-wave activation rows (row = 4p + d; d==3 -> zero pad)
    float* S = S1[wid];
    for (int i = lane; i < 64 * 32; i += 32) {
        int r = i >> 5, c = i & 31;
        int p = r >> 2, d = r & 3, n = n0w + p;
        S[r * TS1 + c] = (d < 3 && n < N) ? featsT[n * 96 + d * 32 + c] : 0.f;
    }
    __syncthreads();

    // GEMM1 (all tiles first: output cols alias input channels, so the full
    // read must finish before the in-place vn_act writes)
    v8f g[4][2];
    #pragma unroll
    for (int mt = 0; mt < 4; ++mt)
        #pragma unroll
        for (int nt = 0; nt < 2; ++nt)
            g[mt][nt] = wmma_tile_k32(S, TS1, mt * 16, Wr, TW1, nt * 16);

    // vn_act on feats, in place (each (row,col) owned by exactly one lane)
    #pragma unroll
    for (int mt = 0; mt < 4; ++mt)
        #pragma unroll
        for (int nt = 0; nt < 2; ++nt) {
            int col = nt * 16 + l;
            #pragma unroll
            for (int t = 0; t < 2; ++t) {
                int rbase = mt * 16 + 8 * hf + 4 * t;
                float pv[3], dv[3], ov[3];
                #pragma unroll
                for (int d = 0; d < 3; ++d) {
                    pv[d] = S[(rbase + d) * TS1 + col];
                    dv[d] = g[mt][nt][4 * t + d];
                }
                vn_act3(pv, dv, ov);
                #pragma unroll
                for (int d = 0; d < 3; ++d)
                    S[(rbase + d) * TS1 + col] = ov[d];
            }
        }
    __syncthreads();

    // GEMM2 pair + register-resident final vn_act -> global stores
    #pragma unroll
    for (int mt = 0; mt < 4; ++mt)
        #pragma unroll
        for (int nt = 0; nt < 4; ++nt) {
            v8f a = wmma_tile_k32(S, TS1, mt * 16, Wu,  TW2, nt * 16);
            v8f b = wmma_tile_k32(S, TS1, mt * 16, Wdu, TW2, nt * 16);
            int o = nt * 16 + l;
            #pragma unroll
            for (int t = 0; t < 2; ++t) {
                int p = (mt * 16 + 8 * hf + 4 * t) >> 2;
                int n = n0w + p;
                if (n < N) {
                    float pv[3], dv[3], ov[3];
                    #pragma unroll
                    for (int d = 0; d < 3; ++d) {
                        pv[d] = a[4 * t + d] * BNS;
                        dv[d] = b[4 * t + d];
                    }
                    vn_act3(pv, dv, ov);
                    #pragma unroll
                    for (int d = 0; d < 3; ++d)
                        out[n * 192 + o * 3 + d] = ov[d];
                }
            }
        }
}

// ---------------------------------------------------------------------------
extern "C" void kernel_launch(void* const* d_in, const int* in_sizes, int n_in,
                              void* d_out, int out_size, void* d_ws, size_t ws_size,
                              hipStream_t stream)
{
    const float* q_pts   = (const float*)d_in[0];
    const float* s_pts   = (const float*)d_in[1];
    /* d_in[2] = s_feats: unused by the reference */
    const int*   nbr     = (const int*)  d_in[3];
    const float* wb      = (const float*)d_in[4];
    const float* w_vn    = (const float*)d_in[5];
    const float* wd_vn   = (const float*)d_in[6];
    const float* w_h1    = (const float*)d_in[7];
    const float* w_h2    = (const float*)d_in[8];
    const float* b_h2    = (const float*)d_in[9];
    const float* wd_relu = (const float*)d_in[10];
    const float* w_un    = (const float*)d_in[11];
    const float* wd_un   = (const float*)d_in[12];

    const int N = in_sizes[0] / 3;          // 16000
    float* featsT = (float*)d_ws;           // [N][3][32] f32 = 6.1 MB

    feats_kernel<<<(N + 15) / 16, 256, 0, stream>>>(q_pts, s_pts, nbr, w_vn, wd_vn,
                                                    w_h1, w_h2, b_h2, wb, featsT, N);
    tail_kernel<<<(N + 63) / 64, 128, 0, stream>>>(featsT, wd_relu, w_un, wd_un,
                                                   (float*)d_out, N);
}
